// UniPortraitLoRAIPAttnProcessor2_0_52381421142053
// MI455X (gfx1250) — matrix operand order repaired and checked by reading, not compiled, not run on hardware
//
#include <hip/hip_runtime.h>
#include <hip/hip_bf16.h>

// ---------------- problem constants ----------------
#define BATCH   8
#define SEQ     4096
#define HID     1280
#define CROSS   768
#define HEADS   20
#define HDIM    64
#define RANK    128
#define NUM_IP  4
#define NFID    16
#define NFACE   2
#define TEXTL   77
#define ENCL    (TEXTL + NUM_IP + NFACE * NFID)   // 113
#define E2      (2 * HID)                          // 2560
#define C2      (2 * CROSS)                        // 1536
#define IP_SCALE  0.6f
#define FID_SCALE 0.7f
#define MID_LORA  0.6f
#define SCALE     0.125f                           // 1/sqrt(64)

// ---------------- bf16 helpers ----------------
__device__ __forceinline__ unsigned short f2us(float x) {
    unsigned u = __float_as_uint(x);
    unsigned r = u + 0x7FFFu + ((u >> 16) & 1u);   // round-to-nearest-even
    return (unsigned short)(r >> 16);
}
__device__ __forceinline__ float us2f(unsigned short s) {
    return __uint_as_float(((unsigned)s) << 16);
}
__device__ __forceinline__ float bflo(unsigned u) {            // low bf16 of packed u32
    return __uint_as_float(u << 16);
}
__device__ __forceinline__ float bfhi(unsigned u) {            // high bf16 of packed u32
    return __uint_as_float(u & 0xffff0000u);
}
// branchless exact-flavor gelu: erf via Abramowitz–Stegun 7.1.26 (|err| ~ 1.5e-7)
__device__ __forceinline__ float gelu_exact(float x) {
    float ax = fabsf(x) * 0.70710678118654752f;
    float t  = __frcp_rn(1.0f + 0.3275911f * ax);
    float p  = t * (0.254829592f + t * (-0.284496736f + t * (1.421413741f +
               t * (-1.453152027f + t * 1.061405429f))));
    float e  = 1.0f - p * __expf(-ax * ax);
    float erfv = copysignf(e, x);
    return 0.5f * x * (1.0f + erfv);
}

// ---------------- WMMA fragment types ----------------
typedef __attribute__((ext_vector_type(16))) __bf16 v16bf;
typedef __attribute__((ext_vector_type(8)))  float  v8f;

union FragU { v16bf v; uint4 q[2]; };

// A fragment (16x32 MxK) from bf16 row-major: lane%16 = M; K halves at 8*half and 16+8*half.
// Contiguous in memory -> two 16B loads.
__device__ __forceinline__ v16bf frag_a(const unsigned short* __restrict__ base, int ld, int lane) {
    const unsigned short* p = base + (lane & 15) * ld + (lane >> 4) * 8;
    FragU f;
    f.q[0] = *(const uint4*)(p);        // K rel 0..7
    f.q[1] = *(const uint4*)(p + 16);   // K rel 16..23
    return f.v;
}
// B fragment (32x16 KxN) from row-major weight W[n][k]: lane%16 = N; K halves at 8*half and 16+8*half.
__device__ __forceinline__ v16bf frag_b(const unsigned short* __restrict__ base, int ld, int lane) {
    const unsigned short* p = base + (lane & 15) * ld + (lane >> 4) * 8;
    FragU f;
    f.q[0] = *(const uint4*)(p);
    f.q[1] = *(const uint4*)(p + 16);
    return f.v;
}
__device__ __forceinline__ v8f wmma_bf16(v16bf a, v16bf b, v8f c) {
    return __builtin_amdgcn_wmma_f32_16x16x32_bf16(false, a, false, b, (short)0, c, false, false);
}

// ================= conversion / precompute kernels =================

// 4-wide f32 -> bf16 convert
__global__ void cvt_bf16_kernel(const float* __restrict__ in, unsigned short* __restrict__ out, int n4) {
    int i = blockIdx.x * 256 + threadIdx.x;
    if (i >= n4) return;
    float4 v = ((const float4*)in)[i];
    ushort4 o;
    o.x = f2us(v.x); o.y = f2us(v.y); o.z = f2us(v.z); o.w = f2us(v.w);
    ((ushort4*)out)[i] = o;
}

// out_bf16[n*K+k] = bf16( wbase[n*K+k] + 0.6 * sum_r lu[n*RANK+r] * ld[r*K+k] )
__global__ void eff_weight_bf16_kernel(const float* __restrict__ wbase, const float* __restrict__ lu,
                                       const float* __restrict__ ld, unsigned short* __restrict__ out,
                                       int N, int K) {
    int i = blockIdx.x * 256 + threadIdx.x;
    if (i >= N * K) return;
    int nn = i / K, kk = i - nn * K;
    float s = 0.f;
    for (int r = 0; r < RANK; ++r) s += lu[nn * RANK + r] * ld[r * K + kk];
    out[i] = f2us(wbase[i] + MID_LORA * s);
}

// f32 effective weight (for small K/V projections)
__global__ void eff_weight_kernel(const float* __restrict__ wbase, const float* __restrict__ lu,
                                  const float* __restrict__ ld, float* __restrict__ out, int N, int K) {
    int i = blockIdx.x * 256 + threadIdx.x;
    if (i >= N * K) return;
    int nn = i / K, kk = i - nn * K;
    float s = 0.f;
    for (int r = 0; r < RANK; ++r) s += lu[nn * RANK + r] * ld[r * K + kk];
    out[i] = wbase[i] + MID_LORA * s;
}

// y[((b*L+t)*HID)+n] = enc[b, roff+t, :] . W[n, :]   (K = CROSS, float4 vectorized)
__global__ void lin_enc_kernel(const float* __restrict__ enc, int roff, int L,
                               const float* __restrict__ W, float* __restrict__ y) {
    int i = blockIdx.x * 256 + threadIdx.x;
    if (i >= BATCH * L * HID) return;
    int nn = i % HID;
    int t  = (i / HID) % L;
    int b  = i / (HID * L);
    const float4* x  = (const float4*)(enc + (size_t)(b * ENCL + roff + t) * CROSS);
    const float4* wr = (const float4*)(W + (size_t)nn * CROSS);
    float s = 0.f;
    for (int k = 0; k < CROSS / 4; ++k) {
        float4 a = x[k], w4 = wr[k];
        s += a.x * w4.x + a.y * w4.y + a.z * w4.z + a.w * w4.w;
    }
    y[i] = s;
}

// r_agg[b][f][c]
__global__ void ragg_kernel(const float* __restrict__ enc, const float* __restrict__ aggr_w,
                            const float* __restrict__ aggr_b, float* __restrict__ ragg) {
    int i = blockIdx.x * 256 + threadIdx.x;
    if (i >= BATCH * NFACE * CROSS) return;
    int c = i % CROSS;
    int f = (i / CROSS) % NFACE;
    int b = i / (CROSS * NFACE);
    float s = 0.f;
    for (int t = 0; t < NFID; ++t)
        s += enc[(size_t)(b * ENCL + TEXTL + NUM_IP + f * NFID + t) * CROSS + c] * aggr_w[t];
    ragg[i] = s + aggr_b[0];
}

__global__ void kr1_kernel(const float* __restrict__ ragg, const float* __restrict__ w1,
                           const float* __restrict__ b1, float* __restrict__ g2) {
    int i = blockIdx.x * 256 + threadIdx.x;
    if (i >= BATCH * NFACE * C2) return;
    int e = i % C2, bf = i / C2;
    float s = 0.f;
    for (int c = 0; c < CROSS; ++c) s += ragg[bf * CROSS + c] * w1[(size_t)e * CROSS + c];
    g2[i] = gelu_exact(s + b1[e]);
}

__global__ void kr2_kernel(const float* __restrict__ g2, const float* __restrict__ w2,
                           float* __restrict__ rk) {
    int i = blockIdx.x * 256 + threadIdx.x;
    if (i >= BATCH * NFACE * HID) return;
    int d = i % HID, bf = i / HID;
    float s = 0.f;
    for (int e = 0; e < C2; ++e) s += g2[bf * C2 + e] * w2[(size_t)d * C2 + e];
    rk[i] = s;
}

__global__ void u_kernel(const float* __restrict__ rk, const float* __restrict__ qw2,
                         float* __restrict__ u) {
    int i = blockIdx.x * 256 + threadIdx.x;
    if (i >= BATCH * NFACE * E2) return;
    int e = i % E2, bf = i / E2;
    float s = 0.f;
    for (int d = 0; d < HID; ++d) s += rk[bf * HID + d] * qw2[(size_t)d * E2 + e];
    u[i] = s;
}

// ================= WMMA kernels (all-bf16 operands) =================

// Routing: logits[row,f] = sum_e gelu(h·qr_w1[e] + b1[e]) * u[b,f,e] -> argmax
__global__ __launch_bounds__(256) void route_kernel(const unsigned short* __restrict__ hsb,
                                                    const unsigned short* __restrict__ w1b,
                                                    const float* __restrict__ b1,
                                                    const float* __restrict__ u,
                                                    int* __restrict__ route) {
    int w = threadIdx.x >> 5, lane = threadIdx.x & 31;
    int row0 = blockIdx.x * 128 + w * 16;        // flattened B*S row
    int b = row0 / SEQ;
    int half = lane >> 4, n = lane & 15;
    float l0[8], l1[8];
#pragma unroll
    for (int r = 0; r < 8; ++r) { l0[r] = 0.f; l1[r] = 0.f; }
    const float* u0p = u + (size_t)(b * NFACE + 0) * E2;
    const float* u1p = u + (size_t)(b * NFACE + 1) * E2;
    const unsigned short* abase = hsb + (size_t)row0 * HID;

    for (int e0 = 0; e0 < E2; e0 += 64) {
        v8f acc[4];
#pragma unroll
        for (int t = 0; t < 4; ++t) acc[t] = (v8f){0.f,0.f,0.f,0.f,0.f,0.f,0.f,0.f};
        for (int kk = 0; kk < HID; kk += 32) {
            __builtin_prefetch(abase + kk + 128, 0, 1);
            v16bf af = frag_a(abase + kk, HID, lane);
#pragma unroll
            for (int t = 0; t < 4; ++t) {
                v16bf bf = frag_b(w1b + (size_t)(e0 + t * 16) * HID + kk, HID, lane);
                acc[t] = wmma_bf16(af, bf, acc[t]);
            }
        }
#pragma unroll
        for (int t = 0; t < 4; ++t) {
            int e = e0 + t * 16 + n;
            float bia = b1[e];
            float uv0 = u0p[e], uv1 = u1p[e];
#pragma unroll
            for (int r = 0; r < 8; ++r) {
                float g = gelu_exact(acc[t][r] + bia);
                l0[r] += g * uv0;
                l1[r] += g * uv1;
            }
        }
    }
#pragma unroll
    for (int r = 0; r < 8; ++r) {
        for (int mk = 1; mk < 16; mk <<= 1) {
            l0[r] += __shfl_xor(l0[r], mk, 32);
            l1[r] += __shfl_xor(l1[r], mk, 32);
        }
    }
    if (n == 0) {
#pragma unroll
        for (int r = 0; r < 8; ++r) {
            int row = row0 + r + 8 * half;
            route[row] = (l1[r] > l0[r]) ? 1 : 0;
        }
    }
}

// query(bf16) = hsb(bf16) @ wq_eff^T(bf16)
__global__ __launch_bounds__(256) void qproj_kernel(const unsigned short* __restrict__ A,
                                                    const unsigned short* __restrict__ W,
                                                    unsigned short* __restrict__ out) {
    int w = threadIdx.x >> 5, lane = threadIdx.x & 31;
    int row0 = blockIdx.x * 128 + w * 16;
    int col0 = blockIdx.y * 64;
    int half = lane >> 4, n = lane & 15;
    v8f acc[4];
#pragma unroll
    for (int t = 0; t < 4; ++t) acc[t] = (v8f){0.f,0.f,0.f,0.f,0.f,0.f,0.f,0.f};
    const unsigned short* abase = A + (size_t)row0 * HID;
    for (int kk = 0; kk < HID; kk += 32) {
        __builtin_prefetch(abase + kk + 128, 0, 1);
        v16bf af = frag_a(abase + kk, HID, lane);
#pragma unroll
        for (int t = 0; t < 4; ++t) {
            v16bf bf = frag_b(W + (size_t)(col0 + t * 16) * HID + kk, HID, lane);
            acc[t] = wmma_bf16(af, bf, acc[t]);
        }
    }
#pragma unroll
    for (int t = 0; t < 4; ++t) {
        int col = col0 + t * 16 + n;
#pragma unroll
        for (int r = 0; r < 8; ++r) {
            int row = row0 + r + 8 * half;
            out[(size_t)row * HID + col] = f2us(acc[t][r]);
        }
    }
}

// out(f32) = hbuf(bf16) @ wo_eff^T(bf16) + bo + residual
__global__ __launch_bounds__(256) void oproj_kernel(const unsigned short* __restrict__ A,
                                                    const unsigned short* __restrict__ W,
                                                    const float* __restrict__ bo,
                                                    const float* __restrict__ resid,
                                                    float* __restrict__ out) {
    int w = threadIdx.x >> 5, lane = threadIdx.x & 31;
    int row0 = blockIdx.x * 128 + w * 16;
    int col0 = blockIdx.y * 64;
    int half = lane >> 4, n = lane & 15;
    v8f acc[4];
#pragma unroll
    for (int t = 0; t < 4; ++t) acc[t] = (v8f){0.f,0.f,0.f,0.f,0.f,0.f,0.f,0.f};
    const unsigned short* abase = A + (size_t)row0 * HID;
    for (int kk = 0; kk < HID; kk += 32) {
        __builtin_prefetch(abase + kk + 128, 0, 1);
        v16bf af = frag_a(abase + kk, HID, lane);
#pragma unroll
        for (int t = 0; t < 4; ++t) {
            v16bf bf = frag_b(W + (size_t)(col0 + t * 16) * HID + kk, HID, lane);
            acc[t] = wmma_bf16(af, bf, acc[t]);
        }
    }
#pragma unroll
    for (int t = 0; t < 4; ++t) {
        int col = col0 + t * 16 + n;
        float bia = bo[col];
#pragma unroll
        for (int r = 0; r < 8; ++r) {
            int row = row0 + r + 8 * half;
            size_t idx = (size_t)row * HID + col;
            out[idx] = acc[t][r] + bia + resid[idx];
        }
    }
}

// ================= fused attention kernel =================

template <int NK>
__device__ __forceinline__ void attn_segment(const float* __restrict__ qrow,
                                             const unsigned short* __restrict__ Kseg,
                                             const unsigned short* __restrict__ Vseg,
                                             float* __restrict__ sc, int lane, int d0,
                                             float& o0, float& o1) {
    constexpr int CNT = (NK + 31) / 32;
    float m = -3.0e38f;
    float scv[CNT];
#pragma unroll
    for (int i = 0; i < CNT; ++i) {
        int k = lane + 32 * i;
        float a = 0.f;
        if (k < NK) {
            const unsigned* kp = (const unsigned*)(Kseg + k * HDIM);
#pragma unroll
            for (int d2 = 0; d2 < HDIM / 2; ++d2) {
                unsigned uk = kp[d2];
                a += qrow[2 * d2] * bflo(uk) + qrow[2 * d2 + 1] * bfhi(uk);
            }
            a *= SCALE;
            m = fmaxf(m, a);
        }
        scv[i] = a;
    }
    for (int off = 16; off; off >>= 1) m = fmaxf(m, __shfl_xor(m, off, 32));
    float ls = 0.f;
#pragma unroll
    for (int i = 0; i < CNT; ++i) {
        int k = lane + 32 * i;
        if (k < NK) {
            float e = __expf(scv[i] - m);
            sc[k] = e;
            ls += e;
        }
    }
    for (int off = 16; off; off >>= 1) ls += __shfl_xor(ls, off, 32);
    __builtin_amdgcn_wave_barrier();
    asm volatile("s_wait_dscnt 0" ::: "memory");    // wave-synchronous LDS exchange
    float a0 = 0.f, a1 = 0.f;
    for (int k = 0; k < NK; ++k) {
        float p = sc[k];
        unsigned uv = *(const unsigned*)(Vseg + k * HDIM + d0);   // d0 even -> aligned u32
        a0 += p * bflo(uv);
        a1 += p * bfhi(uv);
    }
    float inv = 1.0f / ls;
    o0 = a0 * inv;
    o1 = a1 * inv;
    __builtin_amdgcn_wave_barrier();
    asm volatile("s_wait_dscnt 0" ::: "memory");
}

__global__ __launch_bounds__(256) void attn_kernel(const unsigned short* __restrict__ qry,
                                                   const float* __restrict__ kt, const float* __restrict__ vt,
                                                   const float* __restrict__ kip, const float* __restrict__ vip,
                                                   const float* __restrict__ kf, const float* __restrict__ vf,
                                                   const int* __restrict__ route,
                                                   unsigned short* __restrict__ hbuf) {
    __shared__ __align__(16) unsigned short sK[TEXTL * HDIM], sV[TEXTL * HDIM];
    __shared__ __align__(16) unsigned short sKi[NUM_IP * HDIM], sVi[NUM_IP * HDIM];
    __shared__ __align__(16) unsigned short sKf[NFACE * NFID * HDIM], sVf[NFACE * NFID * HDIM];
    __shared__ __align__(16) float sQ[64 * HDIM];
    __shared__ __align__(16) float sSc[8 * 128];

    int tile = blockIdx.x & 63;           // S/64 tiles
    int bh   = blockIdx.x >> 6;
    int b = bh / HEADS, hh = bh % HEADS;
    int tid = threadIdx.x;
    int hof = hh * HDIM;

    for (int i = tid; i < TEXTL * HDIM; i += 256) {
        int t = i >> 6, d = i & 63;
        size_t g = (size_t)(b * TEXTL + t) * HID + hof + d;
        sK[i] = f2us(kt[g]);
        sV[i] = f2us(vt[g]);
    }
    for (int i = tid; i < NUM_IP * HDIM; i += 256) {
        int t = i >> 6, d = i & 63;
        size_t g = (size_t)(b * NUM_IP + t) * HID + hof + d;
        sKi[i] = f2us(kip[g]);
        sVi[i] = f2us(vip[g]);
    }
    for (int i = tid; i < NFACE * NFID * HDIM; i += 256) {
        int t = i >> 6, d = i & 63;
        size_t g = (size_t)(b * NFACE * NFID + t) * HID + hof + d;
        sKf[i] = f2us(kf[g]);
        sVf[i] = f2us(vf[g]);
    }
    // q rows: packed u32 loads (two bf16 each)
    for (int i2 = tid; i2 < 64 * (HDIM / 2); i2 += 256) {
        int r = i2 / (HDIM / 2), d2 = i2 % (HDIM / 2);
        int s = tile * 64 + r;
        unsigned uq = *(const unsigned*)(qry + (size_t)(b * SEQ + s) * HID + hof + 2 * d2);
        sQ[r * HDIM + 2 * d2]     = bflo(uq);
        sQ[r * HDIM + 2 * d2 + 1] = bfhi(uq);
    }
    __syncthreads();

    int wave = tid >> 5, lane = tid & 31;
    float* sc = sSc + wave * 128;
    int d0 = lane * 2;

    for (int rr = 0; rr < 8; ++rr) {
        int rl = wave * 8 + rr;
        int s = tile * 64 + rl;
        const float* qrow = sQ + rl * HDIM;
        int face = route[b * SEQ + s];

        float b0, b1v, i0, i1, f0, f1;
        attn_segment<TEXTL>(qrow, sK, sV, sc, lane, d0, b0, b1v);
        attn_segment<NUM_IP>(qrow, sKi, sVi, sc, lane, d0, i0, i1);
        attn_segment<NFID>(qrow, sKf + face * NFID * HDIM, sVf + face * NFID * HDIM,
                           sc, lane, d0, f0, f1);

        float h0 = b0  + IP_SCALE * i0 + FID_SCALE * f0;
        float h1 = b1v + IP_SCALE * i1 + FID_SCALE * f1;
        unsigned short* hp = hbuf + (size_t)(b * SEQ + s) * HID + hof + d0;
        hp[0] = f2us(h0);
        hp[1] = f2us(h1);
    }
}

// ================= launch =================
extern "C" void kernel_launch(void* const* d_in, const int* in_sizes, int n_in,
                              void* d_out, int out_size, void* d_ws, size_t ws_size,
                              hipStream_t stream) {
    (void)in_sizes; (void)n_in; (void)out_size; (void)ws_size;
    const float* hs     = (const float*)d_in[0];
    const float* enc    = (const float*)d_in[1];
    const float* wq     = (const float*)d_in[2];
    const float* wk     = (const float*)d_in[3];
    const float* wv     = (const float*)d_in[4];
    const float* wo     = (const float*)d_in[5];
    const float* bo     = (const float*)d_in[6];
    const float* q_ld   = (const float*)d_in[7];
    const float* q_lu   = (const float*)d_in[8];
    const float* k_ld   = (const float*)d_in[9];
    const float* k_lu   = (const float*)d_in[10];
    const float* v_ld   = (const float*)d_in[11];
    const float* v_lu   = (const float*)d_in[12];
    const float* o_ld   = (const float*)d_in[13];
    const float* o_lu   = (const float*)d_in[14];
    const float* wk_ip  = (const float*)d_in[15];
    const float* wv_ip  = (const float*)d_in[16];
    const float* wk_fid = (const float*)d_in[17];
    const float* wv_fid = (const float*)d_in[18];
    const float* qr_w1  = (const float*)d_in[19];
    const float* qr_b1  = (const float*)d_in[20];
    const float* qr_w2  = (const float*)d_in[21];
    const float* kr_w1  = (const float*)d_in[22];
    const float* kr_b1  = (const float*)d_in[23];
    const float* kr_w2  = (const float*)d_in[24];
    const float* aggr_w = (const float*)d_in[25];
    const float* aggr_b = (const float*)d_in[26];

    // ---- workspace carve ----
    float* ws = (float*)d_ws;
    float* WKE = ws;  ws += (size_t)HID * CROSS;        // f32 eff weights (small projections)
    float* WVE = ws;  ws += (size_t)HID * CROSS;
    float* KT  = ws;  ws += (size_t)BATCH * TEXTL * HID;
    float* VT  = ws;  ws += (size_t)BATCH * TEXTL * HID;
    float* KIP = ws;  ws += (size_t)BATCH * NUM_IP * HID;
    float* VIP = ws;  ws += (size_t)BATCH * NUM_IP * HID;
    float* KF  = ws;  ws += (size_t)BATCH * NFACE * NFID * HID;
    float* VF  = ws;  ws += (size_t)BATCH * NFACE * NFID * HID;
    float* RAGG = ws; ws += (size_t)BATCH * NFACE * CROSS;
    float* G2   = ws; ws += (size_t)BATCH * NFACE * C2;
    float* RK   = ws; ws += (size_t)BATCH * NFACE * HID;
    float* U    = ws; ws += (size_t)BATCH * NFACE * E2;
    int* ROUTE = (int*)ws; ws += (size_t)BATCH * SEQ;
    unsigned short* HSB  = (unsigned short*)ws; ws += ((size_t)BATCH * SEQ * HID) / 2; // hs in bf16
    unsigned short* W1B  = (unsigned short*)ws; ws += ((size_t)E2 * HID) / 2;          // qr_w1 bf16
    unsigned short* WQEB = (unsigned short*)ws; ws += ((size_t)HID * HID) / 2;         // wq_eff bf16
    unsigned short* WOEB = (unsigned short*)ws; ws += ((size_t)HID * HID) / 2;         // wo_eff bf16
    unsigned short* QRY  = (unsigned short*)ws; ws += ((size_t)BATCH * SEQ * HID) / 2;
    unsigned short* HB   = HSB;   // alias: hs-bf16 is dead after qproj; reuse for attention output

    const int T = 256;
    // bf16 conversions
    cvt_bf16_kernel<<<((BATCH * SEQ * HID / 4) + T - 1) / T, T, 0, stream>>>(hs, HSB, BATCH * SEQ * HID / 4);
    cvt_bf16_kernel<<<((E2 * HID / 4) + T - 1) / T, T, 0, stream>>>(qr_w1, W1B, E2 * HID / 4);
    // LoRA folds
    eff_weight_bf16_kernel<<<(HID * HID + T - 1) / T, T, 0, stream>>>(wq, q_lu, q_ld, WQEB, HID, HID);
    eff_weight_bf16_kernel<<<(HID * HID + T - 1) / T, T, 0, stream>>>(wo, o_lu, o_ld, WOEB, HID, HID);
    eff_weight_kernel<<<(HID * CROSS + T - 1) / T, T, 0, stream>>>(wk, k_lu, k_ld, WKE, HID, CROSS);
    eff_weight_kernel<<<(HID * CROSS + T - 1) / T, T, 0, stream>>>(wv, v_lu, v_ld, WVE, HID, CROSS);
    // router chain (tiny)
    ragg_kernel<<<(BATCH * NFACE * CROSS + T - 1) / T, T, 0, stream>>>(enc, aggr_w, aggr_b, RAGG);
    kr1_kernel<<<(BATCH * NFACE * C2 + T - 1) / T, T, 0, stream>>>(RAGG, kr_w1, kr_b1, G2);
    kr2_kernel<<<(BATCH * NFACE * HID + T - 1) / T, T, 0, stream>>>(G2, kr_w2, RK);
    u_kernel<<<(BATCH * NFACE * E2 + T - 1) / T, T, 0, stream>>>(RK, qr_w2, U);
    // small K/V projections
    lin_enc_kernel<<<(BATCH * TEXTL * HID + T - 1) / T, T, 0, stream>>>(enc, 0, TEXTL, WKE, KT);
    lin_enc_kernel<<<(BATCH * TEXTL * HID + T - 1) / T, T, 0, stream>>>(enc, 0, TEXTL, WVE, VT);
    lin_enc_kernel<<<(BATCH * NUM_IP * HID + T - 1) / T, T, 0, stream>>>(enc, TEXTL, NUM_IP, wk_ip, KIP);
    lin_enc_kernel<<<(BATCH * NUM_IP * HID + T - 1) / T, T, 0, stream>>>(enc, TEXTL, NUM_IP, wv_ip, VIP);
    lin_enc_kernel<<<(BATCH * NFACE * NFID * HID + T - 1) / T, T, 0, stream>>>(enc, TEXTL + NUM_IP, NFACE * NFID, wk_fid, KF);
    lin_enc_kernel<<<(BATCH * NFACE * NFID * HID + T - 1) / T, T, 0, stream>>>(enc, TEXTL + NUM_IP, NFACE * NFID, wv_fid, VF);
    // routing (fused WMMA GEMM + gelu + dual dot + argmax)
    route_kernel<<<(BATCH * SEQ) / 128, T, 0, stream>>>(HSB, W1B, qr_b1, U, ROUTE);
    // Q projection (WMMA)
    dim3 gq((BATCH * SEQ) / 128, HID / 64);
    qproj_kernel<<<gq, T, 0, stream>>>(HSB, WQEB, QRY);
    // fused 3-way attention + routing select  (writes HB, which aliases the now-dead HSB)
    attn_kernel<<<BATCH * HEADS * (SEQ / 64), T, 0, stream>>>(QRY, KT, VT, KIP, VIP, KF, VF, ROUTE, HB);
    // output projection + bias + residual (WMMA)
    dim3 go((BATCH * SEQ) / 128, HID / 64);
    oproj_kernel<<<go, T, 0, stream>>>(HB, WOEB, bo, hs, (float*)d_out);
}
